// RNN_60017872995032
// MI455X (gfx1250) — matrix operand order repaired and checked by reading, not compiled, not run on hardware
//
#include <hip/hip_runtime.h>
#include <stdint.h>

typedef __attribute__((ext_vector_type(16))) __bf16  v16bf;
typedef __attribute__((ext_vector_type(8)))  float   v8f;
typedef __attribute__((ext_vector_type(4)))  unsigned u32x4;
typedef __attribute__((ext_vector_type(8)))  int      i32x8;
typedef __attribute__((ext_vector_type(4)))  int      i32x4;

#define B_    32
#define T_    100
#define V_    20000
#define E_    128
#define TH3   384
#define KSEG  5
#define KPER  (V_ / KSEG)      // 4000
#define NCHUNK (KPER / 32)     // 125
#define MROWS 3264             // 51 * 64 (3200 x-rows + 32 a-rows + pad)
#define MX    3200             // B*T

#if __has_builtin(__builtin_amdgcn_tensor_load_to_lds) && \
    __has_builtin(__builtin_amdgcn_s_wait_tensorcnt)
#define HAVE_TDM 1
#else
#define HAVE_TDM 0
#endif

__device__ inline unsigned short f2bf(float f) {
    unsigned u = __float_as_uint(f);
    unsigned r = u + 0x7FFFu + ((u >> 16) & 1u);   // round-to-nearest-even
    return (unsigned short)(r >> 16);
}
__device__ inline unsigned pack2(float a, float b) {
    return (unsigned)f2bf(a) | ((unsigned)f2bf(b) << 16);
}

// ---------------------------------------------------------------------------
// Tiled f32 -> bf16 transpose: src[R][C] f32 -> dst[C][R] bf16
// ---------------------------------------------------------------------------
__global__ void cvt_transpose(const float* __restrict__ src,
                              unsigned short* __restrict__ dst, int R, int C) {
    __shared__ float tile[32][33];
    int c0 = blockIdx.x * 32, r0 = blockIdx.y * 32;
    int tx = threadIdx.x, ty = threadIdx.y;      // 32 x 8
    for (int i = 0; i < 4; ++i) {
        int r = r0 + ty + i * 8, c = c0 + tx;
        float v = 0.f;
        if (r < R && c < C) v = src[(size_t)r * C + c];
        tile[ty + i * 8][tx] = v;
    }
    __syncthreads();
    for (int i = 0; i < 4; ++i) {
        int c = c0 + ty + i * 8, r = r0 + tx;
        if (c < C && r < R) dst[(size_t)c * R + r] = f2bf(tile[tx][ty + i * 8]);
    }
}

#if HAVE_TDM
// ---------------------------------------------------------------------------
// One TDM load of a [16 x 32] f32 tile (row stride V_ elements) into LDS at
// byte offset lds_off. D# per CDNA5 ISA sec 8.3/8.4. Issued per wave: EXEC
// is ignored, TENSORcnt is per-wave and same-wave tensor ops are in-order.
// ---------------------------------------------------------------------------
__device__ inline void tdm_load_tile(const float* gaddr, unsigned lds_off) {
    unsigned long long ga = (unsigned long long)(uintptr_t)gaddr;
    u32x4 g0;
    g0[0] = 1u;                                       // count=1 (user, valid)
    g0[1] = lds_off;                                  // lds_addr (bytes)
    g0[2] = (unsigned)(ga & 0xFFFFFFFFu);             // global_addr[31:0]
    g0[3] = ((unsigned)(ga >> 32) & 0x01FFFFFFu)      // global_addr[56:32]
            | 0x80000000u;                            // type=2 ("image")
    i32x8 g1;
    g1[0] = (int)(2u << 16);                          // data_size = 2 -> 4B
    g1[1] = (int)(32u << 16);                         // tensor_dim0[15:0]=32
    g1[2] = (int)(16u << 16);                         // tensor_dim1[15:0]=16
    g1[3] = (int)(32u << 16);                         // tile_dim0=32
    g1[4] = (int)16;                                  // tile_dim1=16, dim2=0
    g1[5] = (int)V_;                                  // dim0_stride[31:0]=20000
    g1[6] = 0;                                        // stride hi / dim1_stride
    g1[7] = 0;
    i32x4 z4 = {0, 0, 0, 0};
    i32x8 z8 = {0, 0, 0, 0, 0, 0, 0, 0};
    __builtin_amdgcn_tensor_load_to_lds(g0, g1, z4, z4, z8, 0);
}
#endif

// ---------------------------------------------------------------------------
// Big GEMM (partials): rows 0..3199 = x@emb, rows 3200..3231 = a@emb.
// 128 threads = 4 waves; wave w owns rows rb*64 + w*16 .. +15, all 128 cols.
// TDM path: per-wave double-buffered tensor_load_to_lds, depth-2 pipeline,
// NO workgroup barriers (per-wave in-order TENSORcnt/DScnt make it safe).
// Fallback: direct global A-fragment loads (compile-compat only).
// grid = (51 row-blocks, KSEG k-segments).
// ---------------------------------------------------------------------------
__global__ __launch_bounds__(128)
void gemm_xe_partial(const float* __restrict__ x, const float* __restrict__ a,
                     const unsigned short* __restrict__ embT,  // [E_][V_] bf16
                     float* __restrict__ part)                 // [KSEG][MROWS][E_]
{
    const int rb   = blockIdx.x;
    const int ks   = blockIdx.y;
    const int tid  = threadIdx.x;
    const int wave = tid >> 5, lane = tid & 31;
    const int l15  = lane & 15, h = lane >> 4;
    const int k0   = ks * KPER;
    const int rowbase = rb * 64 + wave * 16;

    if (rowbase >= MX + B_) return;   // rb==50, waves 2,3 (kernel is barrier-free)
    const float* gbase = (rowbase < MX) ? (x + (size_t)rowbase * V_)
                                        : (a + (size_t)(rowbase - MX) * V_);

    v8f acc[8];
#pragma unroll
    for (int i = 0; i < 8; ++i)
#pragma unroll
        for (int r = 0; r < 8; ++r) acc[i][r] = 0.f;

#if HAVE_TDM
    __shared__ __align__(16) float ldsA[4][2][16 * 32];   // per-wave 2 buffers
    float* buf0 = &ldsA[wave][0][0];
    float* buf1 = &ldsA[wave][1][0];
    tdm_load_tile(gbase + k0, (unsigned)(uintptr_t)buf0);
#endif

    for (int ch = 0; ch < NCHUNK; ++ch) {
        const int kg = k0 + ch * 32;
#if HAVE_TDM
        const int cur = ch & 1;
        if (ch + 1 < NCHUNK) {
            // overlap: DMA chunk ch+1 into the other buffer, wait only for ch
            tdm_load_tile(gbase + kg + 32,
                          (unsigned)(uintptr_t)(cur ? buf0 : buf1));
            __builtin_amdgcn_s_wait_tensorcnt(1);
        } else {
            __builtin_amdgcn_s_wait_tensorcnt(0);
        }
        const float* arow = (cur ? buf1 : buf0) + l15 * 32;
#else
        const float* arow = gbase + (size_t)l15 * V_ + kg;
#endif
        // A fragment: f32 -> bf16 (ISA A layout: v0-3: K=8h+[0..7],
        // v4-7: K=16+8h+[0..7])
        float4 fa = *(const float4*)(arow + 8 * h);
        float4 fb = *(const float4*)(arow + 8 * h + 4);
        float4 fc = *(const float4*)(arow + 16 + 8 * h);
        float4 fd = *(const float4*)(arow + 20 + 8 * h);
        union { v16bf v; unsigned u[8]; } af;
        af.u[0] = pack2(fa.x, fa.y); af.u[1] = pack2(fa.z, fa.w);
        af.u[2] = pack2(fb.x, fb.y); af.u[3] = pack2(fb.z, fb.w);
        af.u[4] = pack2(fc.x, fc.y); af.u[5] = pack2(fc.z, fc.w);
        af.u[6] = pack2(fd.x, fd.y); af.u[7] = pack2(fd.z, fd.w);

#pragma unroll
        for (int ct = 0; ct < 8; ++ct) {
            int col = ct * 16 + l15;    // output column (E dim)
            union { v16bf v; u32x4 q[2]; } bf;
            const unsigned char* bp =
                (const unsigned char*)embT + ((size_t)col * V_ + kg + 8 * h) * 2;
            bf.q[0] = *(const u32x4*)bp;
            bf.q[1] = *(const u32x4*)(bp + 32);
            acc[ct] = __builtin_amdgcn_wmma_f32_16x16x32_bf16(
                false, af.v, false, bf.v, (short)0, acc[ct], false, false);
        }
    }

    float* outp = part + (size_t)ks * MROWS * E_ + (size_t)rowbase * E_;
#pragma unroll
    for (int ct = 0; ct < 8; ++ct) {
        int col = ct * 16 + l15;
#pragma unroll
        for (int r = 0; r < 8; ++r)
            outp[(size_t)(r + 8 * h) * E_ + col] = acc[ct][r];
    }
}

// ---------------------------------------------------------------------------
// K-split reduce + tanh + mask + bf16 copy. grid = 3232 rows, 128 threads.
// ---------------------------------------------------------------------------
__global__ void reduce_tanh(const float* __restrict__ part,
                            float* __restrict__ xe_all,
                            unsigned short* __restrict__ xe16,
                            float* __restrict__ mask) {
    __shared__ int anyflag;
    int m = blockIdx.x, e = threadIdx.x;
    if (e == 0) anyflag = 0;
    __syncthreads();
    float s = 0.f;
    for (int ks = 0; ks < KSEG; ++ks)
        s += part[(size_t)ks * MROWS * E_ + (size_t)m * E_ + e];
    float v = tanhf(s);
    xe_all[(size_t)m * E_ + e] = v;
    if (m < MX) {
        xe16[(size_t)m * E_ + e] = f2bf(v);
        if (v != 0.f) anyflag = 1;          // benign same-value race
    }
    __syncthreads();
    if (e == 0 && m < MX) mask[m] = anyflag ? 1.f : 0.f;
}

// ---------------------------------------------------------------------------
// x_proj = xe @ gru_k : [3200x128]x[128x384]. All operands L2-resident,
// fragments loaded straight from global. grid = (50, 3), 256 threads.
// ---------------------------------------------------------------------------
__global__ __launch_bounds__(256)
void gemm_xproj(const unsigned short* __restrict__ xe16,  // [MX][128] bf16
                const unsigned short* __restrict__ gkT,   // [384][128] bf16
                float* __restrict__ xp)                   // [MX][384]
{
    const int rb = blockIdx.x, nb = blockIdx.y;
    const int tid = threadIdx.x, wave = tid >> 5, lane = tid & 31;
    const int rt = wave & 3, cg = wave >> 2;
    const int l15 = lane & 15, h = lane >> 4;
    const int row = rb * 64 + rt * 16 + l15;

    v8f acc[4];
    for (int i = 0; i < 4; ++i)
        for (int r = 0; r < 8; ++r) acc[i][r] = 0.f;

    for (int kc = 0; kc < 4; ++kc) {
        int kg = kc * 32;
        union { v16bf v; u32x4 q[2]; } af;
        const unsigned char* ap =
            (const unsigned char*)xe16 + ((size_t)row * 128 + kg + 8 * h) * 2;
        af.q[0] = *(const u32x4*)ap;
        af.q[1] = *(const u32x4*)(ap + 32);

        for (int ct = 0; ct < 4; ++ct) {
            int col = nb * 128 + cg * 64 + ct * 16 + l15;  // 0..383
            union { v16bf v; u32x4 q[2]; } bf;
            const unsigned char* bp =
                (const unsigned char*)gkT + ((size_t)col * 128 + kg + 8 * h) * 2;
            bf.q[0] = *(const u32x4*)bp;
            bf.q[1] = *(const u32x4*)(bp + 32);
            acc[ct] = __builtin_amdgcn_wmma_f32_16x16x32_bf16(
                false, af.v, false, bf.v, (short)0, acc[ct], false, false);
        }
    }
    for (int ct = 0; ct < 4; ++ct) {
        int col = nb * 128 + cg * 64 + ct * 16 + l15;
        for (int r = 0; r < 8; ++r) {
            int orow = rb * 64 + rt * 16 + r + 8 * h;
            xp[(size_t)orow * TH3 + col] = acc[ct][r];
        }
    }
}

// ---------------------------------------------------------------------------
// Sequential GRU (f32, one WG per batch element) + fused head layers.
// ---------------------------------------------------------------------------
__global__ __launch_bounds__(384)
void gru_head(const float* __restrict__ xp,     // [MX][384]
              const float* __restrict__ mask,   // [MX]
              const float* __restrict__ xe_all, // [MROWS][128]; rows 3200+ = ae
              const float* __restrict__ rk,     // gru_rk [128][384]
              const float* __restrict__ b_i, const float* __restrict__ b_r,
              const float* __restrict__ W1, const float* __restrict__ b1,
              const float* __restrict__ W2, const float* __restrict__ b2,
              const float* __restrict__ d, float* __restrict__ out) {
    __shared__ float h_s[128];
    __shared__ float rec_s[384];
    __shared__ float c_s[64];
    int b = blockIdx.x, j = threadIdx.x;
    if (j < 128) h_s[j] = 0.f;
    __syncthreads();

    for (int t = 0; t < T_; ++t) {
        int m = b * T_ + t;
        float r = b_r[j];
        for (int k = 0; k < 128; ++k) r += h_s[k] * rk[k * TH3 + j];
        rec_s[j] = r;
        __syncthreads();
        if (j < 128) {
            float xz = xp[(size_t)m * TH3 + j]       + b_i[j];
            float xr = xp[(size_t)m * TH3 + 128 + j] + b_i[128 + j];
            float xh = xp[(size_t)m * TH3 + 256 + j] + b_i[256 + j];
            float rz = rec_s[j], rr = rec_s[128 + j], rh = rec_s[256 + j];
            float z  = 1.f / (1.f + expf(-(xz + rz)));
            float rg = 1.f / (1.f + expf(-(xr + rr)));
            float hh = tanhf(xh + rg * rh);
            float hn = z * h_s[j] + (1.f - z) * hh;
            h_s[j] = (mask[m] != 0.f) ? hn : h_s[j];
        }
        __syncthreads();
    }

    if (j < 64) {   // c = tanh([h, ae] @ W1 + b1)
        const float* ae = xe_all + (size_t)(MX + b) * E_;
        float s = b1[j];
        for (int k = 0; k < 128; ++k) s += h_s[k] * W1[k * 64 + j];
        for (int k = 0; k < 128; ++k) s += ae[k] * W1[(128 + k) * 64 + j];
        c_s[j] = tanhf(s);
    }
    __syncthreads();
    if (j == 0) {
        float s = b2[0];
        for (int k = 0; k < 64; ++k) s += c_s[k] * W2[k];
        s += d[b * 2 + 0] * W2[64] + d[b * 2 + 1] * W2[65];
        out[b] = 1.f / (1.f + expf(-s));
    }
}

// ---------------------------------------------------------------------------
extern "C" void kernel_launch(void* const* d_in, const int* in_sizes, int n_in,
                              void* d_out, int out_size, void* d_ws, size_t ws_size,
                              hipStream_t stream) {
    (void)in_sizes; (void)n_in; (void)out_size; (void)ws_size;
    const float* x      = (const float*)d_in[0];
    const float* a      = (const float*)d_in[1];
    const float* d      = (const float*)d_in[2];
    const float* emb    = (const float*)d_in[3];
    const float* gru_k  = (const float*)d_in[4];
    const float* gru_rk = (const float*)d_in[5];
    const float* b_i    = (const float*)d_in[6];
    const float* b_r    = (const float*)d_in[7];
    const float* W1     = (const float*)d_in[8];
    const float* b1     = (const float*)d_in[9];
    const float* W2     = (const float*)d_in[10];
    const float* b2     = (const float*)d_in[11];
    float* out = (float*)d_out;

    char* ws = (char*)d_ws;
    unsigned short* embT  = (unsigned short*)(ws + 0);          // 5,120,000 B
    unsigned short* gkT   = (unsigned short*)(ws + 5120000);    //    98,304 B
    float*          part  = (float*)(ws + 5218304);             // 8,355,840 B
    float*          xeall = (float*)(ws + 13574144);            // 1,671,168 B
    unsigned short* xe16  = (unsigned short*)(ws + 15245312);   //   819,200 B
    float*          xp    = (float*)(ws + 16064512);            // 4,915,200 B
    float*          mask  = (float*)(ws + 20979712);            //    12,800 B

    cvt_transpose<<<dim3(4, 625), dim3(32, 8), 0, stream>>>(emb, embT, V_, E_);
    cvt_transpose<<<dim3(12, 4), dim3(32, 8), 0, stream>>>(gru_k, gkT, E_, TH3);
    gemm_xe_partial<<<dim3(51, KSEG), 128, 0, stream>>>(x, a, embT, part);
    reduce_tanh<<<MX + B_, 128, 0, stream>>>(part, xeall, xe16, mask);
    gemm_xproj<<<dim3(50, 3), 256, 0, stream>>>(xe16, gkT, xp);
    gru_head<<<B_, 384, 0, stream>>>(xp, mask, xeall, gru_rk, b_i, b_r,
                                     W1, b1, W2, b2, d, out);
}
